// GeoLayer_901943132746
// MI455X (gfx1250) — compile-verified, matrix operand order checked by reading, not run on hardware
//
#include <hip/hip_runtime.h>
#include <hip/hip_bf16.h>
#include <math.h>

// ---------------------------------------------------------------------------
// GAT layer for MI455X (gfx1250, wave32).
// Phase 1: h = x @ W via bf16 WMMA (v_wmma_f32_16x16x32_bf16), fp32 accum.
//          B fragments staged in LDS (320 KB/WGP) to decouple WMMA from L2.
// Phase 2: edge softmax + scatter (memory/atomic bound; native f32 atomics,
//          including global_atomic_max_num_f32 via inline CDNA5 asm).
// ---------------------------------------------------------------------------

typedef __attribute__((ext_vector_type(16))) __bf16 v16bf;
typedef __attribute__((ext_vector_type(8)))  float  v8f;

#define IN_F   256
#define OUT_F  128
#define NHEAD  4
#define CHEAD  32
#define NEG_SLOPE 0.2f

#define WFRAG_ELEMS (8 * 8 * 32 * 16)   // tile x ks x lane x 16 bf16 = 64 KB

// -------------------- weight -> bf16 WMMA fragment swizzle -----------------
// Fragment layout for B (32x16 bf16, symmetric to A layout):
//   lane l (<16): column c = l,     K in {0..7, 16..23} within the k-step
//   lane l (>=16): column c = l-16, K in {8..15, 24..31}
__global__ void wprep_kernel(const float* __restrict__ weight,
                             __bf16* __restrict__ wfrag) {
    int t = blockIdx.x * blockDim.x + threadIdx.x;
    if (t >= 8 * 8 * 32) return;
    int lane = t & 31;
    int ks   = (t >> 5) & 7;
    int tile = t >> 8;
    int half = lane >> 4;
    int c    = tile * 16 + (lane & 15);
    __bf16* dst = wfrag + (size_t)t * 16;
#pragma unroll
    for (int e = 0; e < 16; ++e) {
        int kk = ks * 32 + ((e < 8) ? (half * 8 + e) : (16 + half * 8 + (e - 8)));
        dst[e] = (__bf16)weight[kk * OUT_F + c];
    }
}

// -------------------- GEMM: h[N,128] = x[N,256] @ W[256,128] ---------------
// One wave per 16-row strip; 8 output tiles of 16 cols kept in 8 v8f accums.
// K loop: 8 steps of 32; A fragment built once per step and reused 8x.
// All B fragments (64 KB) staged in LDS once per block.
__global__ __launch_bounds__(256) void gemm_wmma_kernel(
        const float* __restrict__ x, const __bf16* __restrict__ wfrag,
        float* __restrict__ h, int N) {
    __shared__ uint4 lds_w[WFRAG_ELEMS / 8];        // 64 KB of bf16 fragments

    // Cooperative stage: global_load_b128 -> ds_store_b128 (16 chunks/thread)
    {
        const uint4* __restrict__ src = (const uint4*)wfrag;
#pragma unroll
        for (int i = 0; i < WFRAG_ELEMS / 8 / 256; ++i) {
            lds_w[i * 256 + threadIdx.x] = src[i * 256 + threadIdx.x];
        }
    }
    __syncthreads();

    const int wave  = threadIdx.x >> 5;
    const int lane  = threadIdx.x & 31;
    const int strip = blockIdx.x * 8 + wave;        // wave-uniform
    const int nstrips = (N + 15) >> 4;
    if (strip >= nstrips) return;                   // whole-wave exit: EXEC all-1s

    const __bf16* __restrict__ lw = (const __bf16*)lds_w;
    const int half = lane >> 4;
    const int r    = lane & 15;
    int row = strip * 16 + r;
    if (row >= N) row = N - 1;                      // clamp (pad rows discarded)
    const float* __restrict__ xrow = x + (size_t)row * IN_F;

    v8f acc[8] = {};

#pragma unroll
    for (int ks = 0; ks < 8; ++ks) {
        const int kb = ks * 32;
        if (ks < 7) __builtin_prefetch(xrow + kb + 32, 0, 3);   // global_prefetch_b8

        // A fragment (16x32 bf16): two 8-wide contiguous K groups per lane.
        const float4 g0a = *(const float4*)(xrow + kb + half * 8);
        const float4 g0b = *(const float4*)(xrow + kb + half * 8 + 4);
        const float4 g1a = *(const float4*)(xrow + kb + 16 + half * 8);
        const float4 g1b = *(const float4*)(xrow + kb + 16 + half * 8 + 4);
        v16bf a;
        a[0]  = (__bf16)g0a.x; a[1]  = (__bf16)g0a.y;
        a[2]  = (__bf16)g0a.z; a[3]  = (__bf16)g0a.w;
        a[4]  = (__bf16)g0b.x; a[5]  = (__bf16)g0b.y;
        a[6]  = (__bf16)g0b.z; a[7]  = (__bf16)g0b.w;
        a[8]  = (__bf16)g1a.x; a[9]  = (__bf16)g1a.y;
        a[10] = (__bf16)g1a.z; a[11] = (__bf16)g1a.w;
        a[12] = (__bf16)g1b.x; a[13] = (__bf16)g1b.y;
        a[14] = (__bf16)g1b.z; a[15] = (__bf16)g1b.w;

#pragma unroll
        for (int t = 0; t < 8; ++t) {
            const v16bf b = *(const v16bf*)(lw + ((((size_t)t * 8 + ks) * 32 + lane) << 4));
            acc[t] = __builtin_amdgcn_wmma_f32_16x16x32_bf16(
                false, a, false, b, (short)0, acc[t], false, false);
        }
    }

    // C/D layout: element e of acc[t] in lane l -> row strip*16 + half*8 + e,
    //             col t*16 + (l & 15).
#pragma unroll
    for (int t = 0; t < 8; ++t) {
        const int c = t * 16 + r;
#pragma unroll
        for (int e = 0; e < 8; ++e) {
            const int rr = strip * 16 + half * 8 + e;
            if (rr < N) h[(size_t)rr * OUT_F + c] = acc[t][e];
        }
    }
}

// -------------------- init: amax=-inf, denom=0; out=bias -------------------
__global__ void init_stats_kernel(float* __restrict__ amax,
                                  float* __restrict__ denom, int n4) {
    int i = blockIdx.x * blockDim.x + threadIdx.x;
    if (i < n4) { amax[i] = -__builtin_inff(); denom[i] = 0.0f; }
}

__global__ void init_out_kernel(float* __restrict__ out,
                                const float* __restrict__ bias, int total) {
    int i = blockIdx.x * blockDim.x + threadIdx.x;
    if (i < total) out[i] = bias[i & (OUT_F - 1)];
}

// -------------------- per-(node,head) attention dots -----------------------
// al[n,h] = h[n,h,:]·att[h,:C]   (applied at dst)
// ar[n,h] = h[n,h,:]·att[h,C:]   (applied at src)
__global__ void node_att_kernel(const float* __restrict__ h,
                                const float* __restrict__ att,
                                float* __restrict__ al, float* __restrict__ ar,
                                int NH) {
    int t = blockIdx.x * blockDim.x + threadIdx.x;
    if (t >= NH) return;
    int n = t >> 2, hh = t & 3;
    const float* __restrict__ hp = h + (size_t)n * OUT_F + hh * CHEAD;
    const float* __restrict__ wl = att + hh * (2 * CHEAD);
    const float* __restrict__ wr = wl + CHEAD;
    float sl = 0.f, sr = 0.f;
#pragma unroll
    for (int c = 0; c < CHEAD; ++c) {
        float v = hp[c];
        sl += v * wl[c];
        sr += v * wr[c];
    }
    al[t] = sl;
    ar[t] = sr;
}

// -------------------- native CDNA5 float atomic max ------------------------
// GLOBAL_ATOMIC_MAX_NUM_F32 without return, device scope (cross-workgroup).
__device__ inline void atomicMaxF(float* addr, float val) {
    asm volatile("global_atomic_max_num_f32 %0, %1, off scope:SCOPE_DEV"
                 :: "v"(addr), "v"(val) : "memory");
}

__device__ inline void edge_sd(const int* __restrict__ ei, int e, int E,
                               int& s, int& d) {
    if (e < E) { s = ei[e]; d = ei[E + e]; }
    else       { s = e - E; d = e - E; }        // appended self loops
}

// -------------------- edge pass 1: raw alpha + segment max (over src) ------
__global__ void edge_max_kernel(const float* __restrict__ al,
                                const float* __restrict__ ar,
                                const int* __restrict__ ei,
                                float* __restrict__ alpha,
                                float* __restrict__ amax,
                                int E, int E2) {
    int t = blockIdx.x * blockDim.x + threadIdx.x;
    if (t >= E2 * NHEAD) return;
    int e = t >> 2, hh = t & 3;
    int s, d; edge_sd(ei, e, E, s, d);
    float a = al[(size_t)d * NHEAD + hh] + ar[(size_t)s * NHEAD + hh];
    a = (a > 0.f) ? a : NEG_SLOPE * a;          // leaky relu
    alpha[t] = a;
    atomicMaxF(&amax[(size_t)s * NHEAD + hh], a);
}

// -------------------- edge pass 2: exp + segment sum (over src) ------------
__global__ void edge_sum_kernel(float* __restrict__ alpha,
                                const float* __restrict__ amax,
                                float* __restrict__ denom,
                                const int* __restrict__ ei,
                                int E, int E2) {
    int t = blockIdx.x * blockDim.x + threadIdx.x;
    if (t >= E2 * NHEAD) return;
    int e = t >> 2, hh = t & 3;
    int s, d; edge_sd(ei, e, E, s, d);
    float a = __expf(alpha[t] - amax[(size_t)s * NHEAD + hh]);
    alpha[t] = a;
    atomicAdd(&denom[(size_t)s * NHEAD + hh], a);
}

// -------------------- edge pass 3: scatter messages into out ---------------
// One wave per edge; lane covers 4 contiguous output floats (float4 load of
// h[src], 4 global_atomic_add_f32 into out[dst]).
__global__ __launch_bounds__(256) void scatter_kernel(
        const float* __restrict__ h, const float* __restrict__ alpha,
        const float* __restrict__ denom, const int* __restrict__ ei,
        float* __restrict__ out, int E, int E2) {
    int warp = threadIdx.x >> 5;
    int lane = threadIdx.x & 31;
    int e = blockIdx.x * 8 + warp;
    if (e >= E2) return;
    int s, d; edge_sd(ei, e, E, s, d);
    int hh = lane >> 3;                          // (lane*4)/32
    float w = alpha[(size_t)e * NHEAD + hh] /
              (denom[(size_t)s * NHEAD + hh] + 1e-16f);
    const float4 v = ((const float4*)(h + (size_t)s * OUT_F))[lane];
    float* op = out + (size_t)d * OUT_F + lane * 4;
    atomicAdd(op + 0, v.x * w);
    atomicAdd(op + 1, v.y * w);
    atomicAdd(op + 2, v.z * w);
    atomicAdd(op + 3, v.w * w);
}

// ---------------------------------------------------------------------------
extern "C" void kernel_launch(void* const* d_in, const int* in_sizes, int n_in,
                              void* d_out, int out_size, void* d_ws, size_t ws_size,
                              hipStream_t stream) {
    const float* x      = (const float*)d_in[0];   // N x 256
    const float* weight = (const float*)d_in[1];   // 256 x 128
    const float* att    = (const float*)d_in[2];   // 1 x 4 x 64
    const float* bias   = (const float*)d_in[3];   // 128
    const int*   ei     = (const int*)d_in[4];     // 2 x E (row0=src, row1=dst)
    float* out = (float*)d_out;

    const int N  = in_sizes[0] / IN_F;
    const int E  = in_sizes[4] / 2;
    const int E2 = E + N;                          // + self loops

    // Workspace layout (floats)
    float* ws    = (float*)d_ws;
    float* h     = ws;                              // N*128
    float* al    = h     + (size_t)N * OUT_F;       // N*4
    float* ar    = al    + (size_t)N * NHEAD;       // N*4
    float* amax  = ar    + (size_t)N * NHEAD;       // N*4
    float* denom = amax  + (size_t)N * NHEAD;       // N*4
    float* alpha = denom + (size_t)N * NHEAD;       // E2*4
    size_t off = (size_t)(alpha - ws) + (size_t)E2 * NHEAD;
    off = (off + 7) & ~(size_t)7;                   // 32B align for v16bf loads
    __bf16* wfrag = (__bf16*)(ws + off);            // 64 KB of fragments

    // 1) weight fragment swizzle (tiny)
    wprep_kernel<<<8, 256, 0, stream>>>(weight, wfrag);

    // 2) WMMA GEMM: one wave per 16-row strip, 8 waves per block
    const int nstrips = (N + 15) / 16;
    gemm_wmma_kernel<<<(nstrips + 7) / 8, 256, 0, stream>>>(x, wfrag, h, N);

    // 3) init stats and output
    init_stats_kernel<<<(N * NHEAD + 255) / 256, 256, 0, stream>>>(amax, denom, N * NHEAD);
    init_out_kernel<<<(N * OUT_F + 255) / 256, 256, 0, stream>>>(out, bias, N * OUT_F);

    // 4) per-node attention dot products
    node_att_kernel<<<(N * NHEAD + 255) / 256, 256, 0, stream>>>(h, att, al, ar, N * NHEAD);

    // 5) edge softmax (max, then exp/sum), segmented by src per reference
    const int EH = E2 * NHEAD;
    edge_max_kernel<<<(EH + 255) / 256, 256, 0, stream>>>(al, ar, ei, alpha, amax, E, E2);
    edge_sum_kernel<<<(EH + 255) / 256, 256, 0, stream>>>(alpha, amax, denom, ei, E, E2);

    // 6) weighted scatter-add into out (aggregated by dst)
    scatter_kernel<<<(E2 + 7) / 8, 256, 0, stream>>>(h, alpha, denom, ei, out, E, E2);
}
